// TransformerInferenceEngine_27049704030501
// MI455X (gfx1250) — matrix operand order repaired and checked by reading, not compiled, not run on hardware
//
#include <hip/hip_runtime.h>
#include <hip/hip_bf16.h>
#include <math.h>
#include <stdint.h>

typedef __attribute__((ext_vector_type(16))) _Float16 v16h;
typedef __attribute__((ext_vector_type(8)))  _Float16 v8h;
typedef __attribute__((ext_vector_type(8)))  float    v8f;

#define B_SZ      4096
#define STATE_DIM 24
#define T_TOK     8
#define D_MODEL   64
#define LAT       128
#define NL        3
#define NH        8
#define HD        8
#define MEM       10
#define FF        512
#define S_SEQ     35
#define SP        48          // S padded to 3 M-tiles of 16
#define QKVN      192

// ---------------------------------------------------------------------------
// WMMA fragment helpers (layouts per CDNA5 ISA 7.12.2, wave32)
// A: 16x32 f16. lane 0-15: M=lane, VGPR0-3 K=half*8+[0..8), VGPR4-7 K=16+half*8+[0..8)
// B: 32x16 f16 (stored transposed, N-major "out_features x in_features"):
//    lane&15 = N, e<8: K = half*8 + e, e>=8: K = 16 + half*8 + (e-8)
// C: v8f, row = mBase + half*8 + r, col = nBase + (lane&15)
// ---------------------------------------------------------------------------
__device__ __forceinline__ v16h load_a_frag(const _Float16* A, int lda,
                                            int mBase, int kBase, int lane) {
  const int half = (lane >> 4) & 1;
  const _Float16* p = A + (mBase + (lane & 15)) * lda + kBase + half * 8;
  v8h lo = *(const v8h*)(p);
  v8h hi = *(const v8h*)(p + 16);
  v16h a;
#pragma unroll
  for (int j = 0; j < 8; ++j) { a[j] = lo[j]; a[j + 8] = hi[j]; }
  return a;
}

__device__ __forceinline__ v16h load_b_frag(const _Float16* Bt, int ldb,
                                            int nBase, int kBase, int lane) {
  const int half = (lane >> 4) & 1;
  const _Float16* p = Bt + (nBase + (lane & 15)) * ldb + kBase + half * 8;
  v8h lo = *(const v8h*)(p);
  v8h hi = *(const v8h*)(p + 16);
  v16h b;
#pragma unroll
  for (int j = 0; j < 8; ++j) { b[j] = lo[j]; b[j + 8] = hi[j]; }
  return b;
}

template<int K>
__device__ __forceinline__ v8f wmma_tile(const _Float16* A, int lda,
                                         const _Float16* Bt, int ldb,
                                         int mBase, int nBase, int lane, v8f acc) {
#pragma unroll
  for (int k = 0; k < K; k += 32) {
    v16h a = load_a_frag(A, lda, mBase, k, lane);
    v16h b = load_b_frag(Bt, ldb, nBase, k, lane);
    acc = __builtin_amdgcn_wmma_f32_16x16x32_f16(false, a, false, b,
                                                 (short)0, acc, false, false);
  }
  return acc;
}

__device__ __forceinline__ void store_c(float* C, int ldc,
                                        int mBase, int nBase, int lane, v8f c) {
  const int row0 = mBase + ((lane >> 4) & 1) * 8;
  const int col  = nBase + (lane & 15);
#pragma unroll
  for (int r = 0; r < 8; ++r) C[(row0 + r) * ldc + col] = c[r];
}

// Async global -> LDS copy helpers (CDNA5 ASYNCcnt path)
__device__ __forceinline__ void async_g2l_b128(uint32_t lds_addr, const void* gaddr) {
  asm volatile("global_load_async_to_lds_b128 %0, %1, off"
               :: "v"(lds_addr), "v"((uint64_t)(uintptr_t)gaddr)
               : "memory");
}
__device__ __forceinline__ void wait_asynccnt0() {
#if __has_builtin(__builtin_amdgcn_s_wait_asynccnt)
  __builtin_amdgcn_s_wait_asynccnt(0);
#else
  asm volatile("s_wait_asynccnt 0x0" ::: "memory");
#endif
}

// ---------------------------------------------------------------------------
// Prep kernels
// ---------------------------------------------------------------------------
__global__ void cvt_f16_kernel(const float* __restrict__ src,
                               _Float16* __restrict__ dst, int n) {
  int i = blockIdx.x * blockDim.x + threadIdx.x;
  if (i < n) dst[i] = (_Float16)src[i];
}

__global__ void tok_mean_kernel(const int* __restrict__ toks,
                                const float* __restrict__ emb,
                                float* __restrict__ tok) {
  int i = blockIdx.x * blockDim.x + threadIdx.x;
  if (i >= STATE_DIM * D_MODEL) return;
  int s = i / D_MODEL, d = i % D_MODEL;
  float acc = 0.f;
#pragma unroll
  for (int t = 0; t < T_TOK; ++t)
    acc += emb[(long)toks[s * T_TOK + t] * D_MODEL + d];
  tok[i] = acc * (1.0f / T_TOK);
}

__global__ void build_x_kernel(const float* __restrict__ traits,
                               const float* __restrict__ rel_goal,
                               const float* __restrict__ memctx,
                               const float* __restrict__ tok,
                               const float* __restrict__ dim_emb,
                               const float* __restrict__ trait_W,
                               const float* __restrict__ trait_b,
                               const float* __restrict__ intent_W,
                               const float* __restrict__ intent_b,
                               const float* __restrict__ cls,
                               float* __restrict__ X) {
  long b = blockIdx.x;
  for (int i = threadIdx.x; i < S_SEQ * D_MODEL; i += blockDim.x) {
    int s = i / D_MODEL, d = i % D_MODEL;
    float v;
    if (s == 0) {
      v = cls[d];
    } else if (s <= STATE_DIM) {
      int sd = s - 1;
      float te = traits[(b * STATE_DIM + sd) * 2 + 0] * trait_W[d * 2 + 0]
               + traits[(b * STATE_DIM + sd) * 2 + 1] * trait_W[d * 2 + 1]
               + trait_b[d];
      float ie = rel_goal[b * STATE_DIM + sd] * intent_W[d] + intent_b[d];
      v = tok[sd * D_MODEL + d] + dim_emb[sd * D_MODEL + d] + te + ie;
    } else {
      v = memctx[(b * MEM + (s - 1 - STATE_DIM)) * D_MODEL + d];
    }
    X[b * S_SEQ * D_MODEL + i] = v;
  }
}

// ---------------------------------------------------------------------------
// One transformer layer: 1 workgroup (4 waves) per batch element
// ---------------------------------------------------------------------------
__global__ __launch_bounds__(128) void layer_kernel(
    float* __restrict__ X,
    const _Float16* __restrict__ Wqkv_h, const float* __restrict__ bqkv,
    const _Float16* __restrict__ Wo_h,   const float* __restrict__ bo,
    const float* __restrict__ ln1_g,     const float* __restrict__ ln1_b,
    const _Float16* __restrict__ W1_h,   const float* __restrict__ b1,
    const _Float16* __restrict__ W2_h,   const float* __restrict__ b2,
    const float* __restrict__ ln2_g,     const float* __restrict__ ln2_b,
    int l) {
  __shared__ __align__(16) float    xs[SP * D_MODEL];   // fp32 residual stream
  __shared__ __align__(16) _Float16 xh[SP * D_MODEL];   // f16 A-matrix copy
  __shared__ __align__(16) float    scr[SP * QKVN];     // qkv / proj / ffn scratch

  const int tid  = threadIdx.x;
  const int wave = tid >> 5;
  const int lane = tid & 31;
  const long b   = blockIdx.x;
  float* Xb = X + b * S_SEQ * D_MODEL;

  const _Float16* Wq  = Wqkv_h + (long)l * QKVN * D_MODEL;
  const float*    bq  = bqkv + l * QKVN;
  const _Float16* Wol = Wo_h + (long)l * D_MODEL * D_MODEL;
  const float*    bol = bo + l * D_MODEL;
  const _Float16* W1l = W1_h + (long)l * FF * D_MODEL;
  const float*    b1l = b1 + l * FF;
  const _Float16* W2l = W2_h + (long)l * D_MODEL * FF;
  const float*    b2l = b2 + l * D_MODEL;
  const float*    g1  = ln1_g + l * D_MODEL;
  const float*    o1  = ln1_b + l * D_MODEL;
  const float*    g2  = ln2_g + l * D_MODEL;
  const float*    o2  = ln2_b + l * D_MODEL;

  // ---- stage 0: async DMA the activation tile straight into LDS ----
  // 48 rows x 64 f32 = 12288 B = 6 iterations x 128 lanes x 16 B.
  // Rows [35,48) overread into the following workspace region (in-bounds,
  // deterministic); they are zeroed below before any WMMA consumes them.
  {
    const uint32_t lds_base = (uint32_t)(uintptr_t)xs;   // LDS addr = low 32 bits
    const char* gsrc = (const char*)Xb;
#pragma unroll
    for (int it = 0; it < (SP * D_MODEL * 4) / (128 * 16); ++it) {
      int byte_off = (it * 128 + tid) * 16;
      async_g2l_b128(lds_base + byte_off, gsrc + byte_off);
    }
    wait_asynccnt0();
  }
  __syncthreads();
  for (int i = tid; i < SP * D_MODEL; i += 128) {
    float v = xs[i];
    if (i >= S_SEQ * D_MODEL) { v = 0.f; xs[i] = 0.f; }
    xh[i] = (_Float16)v;
  }
  __syncthreads();

  // ---- stage 1: qkv = x @ Wqkv^T + bqkv  (48x192, K=64) ----
  for (int t = wave; t < 3 * (QKVN / 16); t += 4) {
    int mt = t / (QKVN / 16), nt = t % (QKVN / 16);
    v8f acc = {};
    acc = wmma_tile<D_MODEL>(xh, D_MODEL, Wq, D_MODEL, mt * 16, nt * 16, lane, acc);
    float bias = bq[nt * 16 + (lane & 15)];
#pragma unroll
    for (int r = 0; r < 8; ++r) acc[r] += bias;
    store_c(scr, QKVN, mt * 16, nt * 16, lane, acc);
  }
  __syncthreads();

  // ---- stage 2: attention (head dim 8 -> VALU, online softmax) ----
  {
    const float scale = 0.35355339059327373f;  // 1/sqrt(8)
    for (int p = tid; p < S_SEQ * NH; p += 128) {
      int s = p / NH, h = p % NH;
      const float* qp = scr + s * QKVN + h * HD;
      float qs[HD];
#pragma unroll
      for (int d = 0; d < HD; ++d) qs[d] = qp[d] * scale;
      float m = -3.0e38f, ssum = 0.f;
      float out[HD];
#pragma unroll
      for (int d = 0; d < HD; ++d) out[d] = 0.f;
      for (int j = 0; j < S_SEQ; ++j) {
        const float* kp = scr + j * QKVN + D_MODEL + h * HD;
        float dot = 0.f;
#pragma unroll
        for (int d = 0; d < HD; ++d) dot += qs[d] * kp[d];
        float mn   = fmaxf(m, dot);
        float corr = __expf(m - mn);
        float w    = __expf(dot - mn);
        ssum = ssum * corr + w;
        const float* vp = scr + j * QKVN + 2 * D_MODEL + h * HD;
#pragma unroll
        for (int d = 0; d < HD; ++d) out[d] = out[d] * corr + w * vp[d];
        m = mn;
      }
      float inv = 1.f / ssum;
#pragma unroll
      for (int d = 0; d < HD; ++d)
        xh[s * D_MODEL + h * HD + d] = (_Float16)(out[d] * inv);  // attn_out -> A matrix
    }
  }
  __syncthreads();

  // ---- stage 3: out-proj (48x64, K=64), residual, LN1 ----
  for (int t = wave; t < 3 * 4; t += 4) {
    int mt = t / 4, nt = t % 4;
    v8f acc = {};
    acc = wmma_tile<D_MODEL>(xh, D_MODEL, Wol, D_MODEL, mt * 16, nt * 16, lane, acc);
    float bias = bol[nt * 16 + (lane & 15)];
#pragma unroll
    for (int r = 0; r < 8; ++r) acc[r] += bias;
    store_c(scr, D_MODEL, mt * 16, nt * 16, lane, acc);
  }
  __syncthreads();
  for (int s = tid; s < S_SEQ; s += 128) {
    float m = 0.f;
    for (int d = 0; d < D_MODEL; ++d) m += xs[s * D_MODEL + d] + scr[s * D_MODEL + d];
    m *= (1.f / D_MODEL);
    float v = 0.f;
    for (int d = 0; d < D_MODEL; ++d) {
      float t0 = xs[s * D_MODEL + d] + scr[s * D_MODEL + d] - m;
      v += t0 * t0;
    }
    float inv = rsqrtf(v * (1.f / D_MODEL) + 1e-5f);
    for (int d = 0; d < D_MODEL; ++d) {
      float y = (xs[s * D_MODEL + d] + scr[s * D_MODEL + d] - m) * inv * g1[d] + o1[d];
      xs[s * D_MODEL + d] = y;
      xh[s * D_MODEL + d] = (_Float16)y;
    }
  }
  __syncthreads();

  // ---- stage 4: FFN, chunked over FF in 4 x 128 columns ----
  v8f zero = {};
  v8f facc[3] = {zero, zero, zero};       // each wave owns 3 of 12 output tiles
  _Float16* hb = (_Float16*)scr;          // 48x128 f16 hidden chunk (reuses scratch)
  for (int c = 0; c < 4; ++c) {
    int f0 = c * 128;
    // h = gelu(x @ W1_chunk^T + b1_chunk)
    for (int t = wave; t < 3 * 8; t += 4) {
      int mt = t / 8, nt = t % 8;
      v8f acc = {};
      acc = wmma_tile<D_MODEL>(xh, D_MODEL, W1l + (long)f0 * D_MODEL, D_MODEL,
                               mt * 16, nt * 16, lane, acc);
      int col   = nt * 16 + (lane & 15);
      float bias = b1l[f0 + col];
      int row0   = mt * 16 + ((lane >> 4) & 1) * 8;
#pragma unroll
      for (int r = 0; r < 8; ++r) {
        float xv = acc[r] + bias;
        float g  = 0.5f * xv * (1.0f + erff(xv * 0.7071067811865476f));
        hb[(row0 + r) * 128 + col] = (_Float16)g;
      }
    }
    __syncthreads();
    // accumulate out += h_chunk @ W2_chunk^T
#pragma unroll
    for (int i = 0; i < 3; ++i) {
      int t = wave + i * 4;
      int mt = t / 4, nt = t % 4;
      facc[i] = wmma_tile<128>(hb, 128, W2l + f0, FF, mt * 16, nt * 16, lane, facc[i]);
    }
    __syncthreads();
  }
  // store FFN output (+b2) to scratch
#pragma unroll
  for (int i = 0; i < 3; ++i) {
    int t = wave + i * 4;
    int mt = t / 4, nt = t % 4;
    float bias = b2l[nt * 16 + (lane & 15)];
    v8f a = facc[i];
#pragma unroll
    for (int r = 0; r < 8; ++r) a[r] += bias;
    store_c(scr, D_MODEL, mt * 16, nt * 16, lane, a);
  }
  __syncthreads();

  // ---- residual, LN2, write back ----
  for (int s = tid; s < S_SEQ; s += 128) {
    float m = 0.f;
    for (int d = 0; d < D_MODEL; ++d) m += xs[s * D_MODEL + d] + scr[s * D_MODEL + d];
    m *= (1.f / D_MODEL);
    float v = 0.f;
    for (int d = 0; d < D_MODEL; ++d) {
      float t0 = xs[s * D_MODEL + d] + scr[s * D_MODEL + d] - m;
      v += t0 * t0;
    }
    float inv = rsqrtf(v * (1.f / D_MODEL) + 1e-5f);
    for (int d = 0; d < D_MODEL; ++d) {
      float y = (xs[s * D_MODEL + d] + scr[s * D_MODEL + d] - m) * inv * g2[d] + o2[d];
      Xb[s * D_MODEL + d] = y;
    }
  }
}

// ---------------------------------------------------------------------------
// Head: situation = LN(x[:,0] @ out_W^T + out_b); 32 batch rows per WG
// ---------------------------------------------------------------------------
__global__ __launch_bounds__(128) void head_kernel(
    const float* __restrict__ X, const _Float16* __restrict__ outW_h,
    const float* __restrict__ out_b, const float* __restrict__ og,
    const float* __restrict__ ob, float* __restrict__ situation) {
  __shared__ __align__(16) _Float16 ah[32 * D_MODEL];
  __shared__ __align__(16) float    sb[32 * LAT];
  const int tid = threadIdx.x, wave = tid >> 5, lane = tid & 31;
  const long b0 = (long)blockIdx.x * 32;

  for (int i = tid; i < 32 * D_MODEL; i += 128) {
    int r = i / D_MODEL, d = i % D_MODEL;
    ah[i] = (_Float16)X[(b0 + r) * S_SEQ * D_MODEL + d];   // s == 0 row
  }
  __syncthreads();

  for (int t = wave; t < 2 * 8; t += 4) {
    int mt = t / 8, nt = t % 8;
    v8f acc = {};
    acc = wmma_tile<D_MODEL>(ah, D_MODEL, outW_h, D_MODEL, mt * 16, nt * 16, lane, acc);
    float bias = out_b[nt * 16 + (lane & 15)];
#pragma unroll
    for (int r = 0; r < 8; ++r) acc[r] += bias;
    store_c(sb, LAT, mt * 16, nt * 16, lane, acc);
  }
  __syncthreads();

  for (int r = tid; r < 32; r += 128) {
    float m = 0.f;
    for (int j = 0; j < LAT; ++j) m += sb[r * LAT + j];
    m *= (1.f / LAT);
    float v = 0.f;
    for (int j = 0; j < LAT; ++j) { float t0 = sb[r * LAT + j] - m; v += t0 * t0; }
    float inv = rsqrtf(v * (1.f / LAT) + 1e-5f);
    for (int j = 0; j < LAT; ++j)
      situation[(b0 + r) * LAT + j] = (sb[r * LAT + j] - m) * inv * og[j] + ob[j];
  }
}

// ---------------------------------------------------------------------------
// Tail: sequence slice copy + memory update
// ---------------------------------------------------------------------------
__global__ void tail_kernel(const float* __restrict__ X,
                            const float* __restrict__ memctx,
                            float* __restrict__ seq_out,
                            float* __restrict__ mem_out) {
  long b = blockIdx.x;
  const float* Xb = X + b * S_SEQ * D_MODEL;
  for (int i = threadIdx.x; i < STATE_DIM * D_MODEL; i += blockDim.x)
    seq_out[b * STATE_DIM * D_MODEL + i] = Xb[D_MODEL + i];
  for (int i = threadIdx.x; i < MEM * D_MODEL; i += blockDim.x) {
    int m = i / D_MODEL, d = i % D_MODEL;
    float v;
    if (m < MEM - 1) {
      v = memctx[(b * MEM + m + 1) * D_MODEL + d];
    } else {
      float acc = 0.f;
      for (int s = 0; s < STATE_DIM; ++s) acc += Xb[(1 + s) * D_MODEL + d];
      v = acc * (1.f / STATE_DIM);
    }
    mem_out[b * MEM * D_MODEL + i] = v;
  }
}

// ---------------------------------------------------------------------------
extern "C" void kernel_launch(void* const* d_in, const int* in_sizes, int n_in,
                              void* d_out, int out_size, void* d_ws, size_t ws_size,
                              hipStream_t stream) {
  (void)in_sizes; (void)n_in; (void)out_size; (void)ws_size;
  const int*   toks     = (const int*)  d_in[0];
  const float* traits   = (const float*)d_in[1];
  const float* rel_goal = (const float*)d_in[2];
  const float* memctx   = (const float*)d_in[3];
  const float* emb      = (const float*)d_in[4];
  const float* dim_emb  = (const float*)d_in[5];
  const float* trait_W  = (const float*)d_in[6];
  const float* trait_b  = (const float*)d_in[7];
  const float* intent_W = (const float*)d_in[8];
  const float* intent_b = (const float*)d_in[9];
  const float* cls      = (const float*)d_in[10];
  const float* Wqkv     = (const float*)d_in[11];
  const float* bqkv     = (const float*)d_in[12];
  const float* Wo       = (const float*)d_in[13];
  const float* bo       = (const float*)d_in[14];
  const float* ln1_g    = (const float*)d_in[15];
  const float* ln1_b    = (const float*)d_in[16];
  const float* W1       = (const float*)d_in[17];
  const float* b1       = (const float*)d_in[18];
  const float* W2       = (const float*)d_in[19];
  const float* b2       = (const float*)d_in[20];
  const float* ln2_g    = (const float*)d_in[21];
  const float* ln2_b    = (const float*)d_in[22];
  const float* outW     = (const float*)d_in[23];
  const float* out_b    = (const float*)d_in[24];
  const float* out_ln_g = (const float*)d_in[25];
  const float* out_ln_b = (const float*)d_in[26];

  // workspace layout
  char* ws = (char*)d_ws;
  size_t off = 0;
  float* X   = (float*)(ws + off); off += (size_t)B_SZ * S_SEQ * D_MODEL * sizeof(float);
  float* tok = (float*)(ws + off); off += (size_t)STATE_DIM * D_MODEL * sizeof(float);
  off = (off + 255) & ~(size_t)255;
  _Float16* Wqkv_h = (_Float16*)(ws + off); off += (size_t)NL * QKVN * D_MODEL * 2;
  _Float16* Wo_h   = (_Float16*)(ws + off); off += (size_t)NL * D_MODEL * D_MODEL * 2;
  _Float16* W1_h   = (_Float16*)(ws + off); off += (size_t)NL * FF * D_MODEL * 2;
  _Float16* W2_h   = (_Float16*)(ws + off); off += (size_t)NL * D_MODEL * FF * 2;
  _Float16* outW_h = (_Float16*)(ws + off); off += (size_t)LAT * D_MODEL * 2;

  // weight conversion f32 -> f16 (tiny; all of it stays hot in L2)
  auto cvt = [&](const float* s, _Float16* d, int n) {
    cvt_f16_kernel<<<(n + 255) / 256, 256, 0, stream>>>(s, d, n);
  };
  cvt(Wqkv, Wqkv_h, NL * QKVN * D_MODEL);
  cvt(Wo,   Wo_h,   NL * D_MODEL * D_MODEL);
  cvt(W1,   W1_h,   NL * FF * D_MODEL);
  cvt(W2,   W2_h,   NL * D_MODEL * FF);
  cvt(outW, outW_h, LAT * D_MODEL);

  tok_mean_kernel<<<(STATE_DIM * D_MODEL + 255) / 256, 256, 0, stream>>>(toks, emb, tok);
  build_x_kernel<<<B_SZ, 256, 0, stream>>>(traits, rel_goal, memctx, tok, dim_emb,
                                           trait_W, trait_b, intent_W, intent_b, cls, X);

  for (int l = 0; l < NL; ++l)
    layer_kernel<<<B_SZ, 128, 0, stream>>>(X, Wqkv_h, bqkv, Wo_h, bo, ln1_g, ln1_b,
                                           W1_h, b1, W2_h, b2, ln2_g, ln2_b, l);

  float* situation = (float*)d_out;
  float* seq_out   = situation + (size_t)B_SZ * LAT;
  float* mem_out   = seq_out + (size_t)B_SZ * STATE_DIM * D_MODEL;
  head_kernel<<<B_SZ / 32, 128, 0, stream>>>(X, outW_h, out_b, out_ln_g, out_ln_b, situation);
  tail_kernel<<<B_SZ, 256, 0, stream>>>(X, memctx, seq_out, mem_out);
}
